// EfficientCrossAttention_62878321214100
// MI455X (gfx1250) — matrix operand order repaired and checked by reading, not compiled, not run on hardware
//
#include <hip/hip_runtime.h>
#include <hip/hip_bf16.h>

// ---------------------------------------------------------------------------
// EfficientCrossAttention for MI455X (gfx1250, wave32, WMMA bf16).
// All matmuls but the scalar emb MLP lower to v_wmma_f32_16x16x32_bf16; epilogues fused.
// Big GEMMs use 32x64 per-wave tiles: 2 A-frags reused over 4 B-frags
// (8 WMMA per 6 b128-fragment loads) to cut L0 traffic per matrix op.
// ---------------------------------------------------------------------------

typedef __attribute__((ext_vector_type(16))) __bf16 v16bf;
typedef __attribute__((ext_vector_type(8)))  float  v8f;
typedef unsigned short u16;

#define CB 32
#define CT 2048
#define CD 512
#define CN 256
#define CL 256
#define CH 8
#define CTE 2048
#define CDH 64
#define CBT (CB * CT)   // 65536

__device__ inline u16 f2bf(float f) {
  union { float f; unsigned u; } x; x.f = f;
  unsigned r = x.u + 0x7FFFu + ((x.u >> 16) & 1u);   // round-to-nearest-even
  return (u16)(r >> 16);
}

union FragBuf { v16bf v; uint4 q[2]; };

// A operand: lane l holds row m=l&15; VGPR0-3 <- K = k0+8*hi .. +7,
// VGPR4-7 <- K = k0+16+8*hi .. +7  (ISA 7.12.2, 16-bit A 16x32)
__device__ inline v16bf load_frag_a(const u16* __restrict__ base, int lda,
                                    int k0, int lane) {
  const int m = lane & 15, hi = lane >> 4;
  const u16* p = base + (long)m * lda + k0 + hi * 8;
  FragBuf f;
  f.q[0] = *(const uint4*)p;
  f.q[1] = *(const uint4*)(p + 16);
  return f.v;
}

// B operand (column-major storage BT[n][k]): lane l holds column n=l&15;
// VGPR0-3 <- K = k0+8*hi .. +7, VGPR4-7 <- K = k0+16+8*hi .. +7 (matches A).
__device__ inline v16bf load_frag_b(const u16* __restrict__ baseT, int ldb,
                                    int k0, int lane) {
  const int n = lane & 15, hi = lane >> 4;
  const u16* p = baseT + (long)n * ldb + k0 + hi * 8;
  FragBuf f;
  f.q[0] = *(const uint4*)p;
  f.q[1] = *(const uint4*)(p + 16);
  return f.v;
}

__device__ inline v8f wmma_bf16(v16bf a, v16bf b, v8f c) {
  return __builtin_amdgcn_wmma_f32_16x16x32_bf16(false, a, false, b,
                                                 (short)0, c, false, false);
}

// ---------------------------------------------------------------------------
// Weight transpose + f32->bf16:  dst[c*R + r] = src[r*C + c]
// ---------------------------------------------------------------------------
__global__ void transpose_bf16(const float* __restrict__ src,
                               u16* __restrict__ dst, int R, int C) {
  long i = (long)blockIdx.x * 256 + threadIdx.x;
  if (i >= (long)R * C) return;
  int r = (int)(i / C), c = (int)(i % C);
  dst[(long)c * R + r] = f2bf(src[i]);
}

// ---------------------------------------------------------------------------
// LayerNorm rows -> bf16. One wave per row, C in {256, 512}.
// ---------------------------------------------------------------------------
__global__ __launch_bounds__(256) void ln_rows(const float* __restrict__ in,
                                               const float* __restrict__ g,
                                               const float* __restrict__ bta,
                                               u16* __restrict__ out, int C) {
  const int lane = threadIdx.x & 31, wave = threadIdx.x >> 5;
  const long row = (long)blockIdx.x * 8 + wave;
  const float* p = in + row * C;
  const int E = C >> 5;
  float v[16];
  float s = 0.f;
  for (int e = 0; e < E; ++e) { v[e] = p[e * 32 + lane]; s += v[e]; }
  for (int d = 1; d < 32; d <<= 1) s += __shfl_xor(s, d, 32);
  const float mean = s / (float)C;
  float ss = 0.f;
  for (int e = 0; e < E; ++e) { float dv = v[e] - mean; ss += dv * dv; }
  for (int d = 1; d < 32; d <<= 1) ss += __shfl_xor(ss, d, 32);
  const float rstd = rsqrtf(ss / (float)C + 1e-5f);
  for (int e = 0; e < E; ++e) {
    int c = e * 32 + lane;
    out[row * C + c] = f2bf((v[e] - mean) * rstd * g[c] + bta[c]);
  }
}

// ---------------------------------------------------------------------------
// emb MLP: eo[b][c] = silu(emb[b]) @ emb_W + emb_b   (32 x 1024, K=2048)
// ---------------------------------------------------------------------------
__global__ __launch_bounds__(256) void emb_gemm(const float* __restrict__ emb,
                                                const float* __restrict__ W,
                                                const float* __restrict__ bb,
                                                float* __restrict__ eo) {
  __shared__ float se[CTE];
  const int b = blockIdx.x, tid = threadIdx.x;
  for (int k = tid; k < CTE; k += 256) {
    float v = emb[(long)b * CTE + k];
    se[k] = v / (1.f + __expf(-v));
  }
  __syncthreads();
  for (int c = tid; c < 2 * CD; c += 256) {
    float acc = bb[c];
#pragma unroll 8
    for (int k = 0; k < CTE; ++k) acc += se[k] * W[(long)k * (2 * CD) + c];
    eo[(long)b * (2 * CD) + c] = acc;
  }
}

// ---------------------------------------------------------------------------
// k / v projection GEMMs (per batch: 256x512, K=256) with fused epilogues.
// mode 0: k = xfn@Wk + bk + (1-tct)*(-1e6), stored transposed f32 (B,512,256)
// mode 1: v = tct*(xfn@Wv) + bv, stored transposed bf16 (B,512,256)
// Block: 64 rows x 512 cols, 32 waves; wave = 16 rows x 64 cols (4 tiles).
// ---------------------------------------------------------------------------
__global__ __launch_bounds__(1024) void kv_gemm(const u16* __restrict__ xfn,
                                                const u16* __restrict__ WT,
                                                const float* __restrict__ bias,
                                                const int* __restrict__ cond,
                                                float* __restrict__ kT,
                                                u16* __restrict__ vT, int mode) {
  const int lane = threadIdx.x & 31, wave = threadIdx.x >> 5;
  const int rowg = wave & 3, colg = wave >> 2;          // colg 0..7
  const int bidx = blockIdx.x >> 2, rb = blockIdx.x & 3;
  const int row0 = rb * 64 + rowg * 16;                 // 0..255 within batch
  const int colbase = colg * 64;
  const u16* A = xfn + ((long)bidx * CN + row0) * CL;
  v8f acc[4];
#pragma unroll
  for (int t = 0; t < 4; ++t)
#pragma unroll
    for (int i = 0; i < 8; ++i) acc[t][i] = 0.f;

  for (int ks = 0; ks < CL / 32; ++ks) {
    const int k0 = ks * 32;
    v16bf a = load_frag_a(A, CL, k0, lane);
#pragma unroll
    for (int t = 0; t < 4; ++t) {
      v16bf b = load_frag_b(WT + (long)(colbase + t * 16) * CL, CL, k0, lane);
      acc[t] = wmma_bf16(a, b, acc[t]);
    }
  }
  const float tct = ((cond[bidx] % 10) > 0) ? 1.f : 0.f;
  const int hi = lane >> 4, nn = lane & 15;
  if (mode == 0) {
    const float maskadd = (1.f - tct) * -1000000.f;
#pragma unroll
    for (int t = 0; t < 4; ++t) {
      const int col = colbase + t * 16 + nn;
      const float bb = bias[col];
#pragma unroll
      for (int i = 0; i < 8; ++i) {
        int row = row0 + i + 8 * hi;
        kT[((long)bidx * CD + col) * CN + row] = acc[t][i] + bb + maskadd;
      }
    }
  } else {
#pragma unroll
    for (int t = 0; t < 4; ++t) {
      const int col = colbase + t * 16 + nn;
      const float bb = bias[col];
#pragma unroll
      for (int i = 0; i < 8; ++i) {
        int row = row0 + i + 8 * hi;
        vT[((long)bidx * CD + col) * CN + row] = f2bf(tct * acc[t][i] + bb);
      }
    }
  }
}

// ---------------------------------------------------------------------------
// Softmax over N (rows of kT, length 256) -> bf16. One wave per row.
// ---------------------------------------------------------------------------
__global__ __launch_bounds__(256) void softmax_rows(const float* __restrict__ in,
                                                    u16* __restrict__ out) {
  const int lane = threadIdx.x & 31, wave = threadIdx.x >> 5;
  const long row = (long)blockIdx.x * 8 + wave;   // B*D = 16384 rows
  const float* p = in + row * CN;
  float v[8];
  float m = -3.4e38f;
  for (int e = 0; e < 8; ++e) { v[e] = p[e * 32 + lane]; m = fmaxf(m, v[e]); }
  for (int d = 1; d < 32; d <<= 1) m = fmaxf(m, __shfl_xor(m, d, 32));
  float s = 0.f;
  for (int e = 0; e < 8; ++e) { v[e] = __expf(v[e] - m); s += v[e]; }
  for (int d = 1; d < 32; d <<= 1) s += __shfl_xor(s, d, 32);
  const float inv = 1.f / s;
  for (int e = 0; e < 8; ++e) out[row * CN + e * 32 + lane] = f2bf(v[e] * inv);
}

// ---------------------------------------------------------------------------
// att[b,h] = k_smT(64xN) @ v(NxDH)  -> stored transposed attT[b,h][l][d] bf16.
// Grid = B*H blocks, 8 waves; wave = 16(d) x 32(l) (2 tiles), K=256.
// ---------------------------------------------------------------------------
__global__ __launch_bounds__(256) void att_gemm(const u16* __restrict__ kTs,
                                                const u16* __restrict__ vT,
                                                u16* __restrict__ attT) {
  const int lane = threadIdx.x & 31, wave = threadIdx.x >> 5;
  const int mt = wave & 3, ng = wave >> 2;          // ng 0..1
  const int bh = blockIdx.x;                        // 0..255
  const int bidx = bh >> 3, head = bh & 7;
  const u16* A = kTs + ((long)bidx * CD + head * CDH + mt * 16) * CN;
  v8f acc[2];
#pragma unroll
  for (int t = 0; t < 2; ++t)
#pragma unroll
    for (int i = 0; i < 8; ++i) acc[t][i] = 0.f;

  for (int ks = 0; ks < CN / 32; ++ks) {
    const int k0 = ks * 32;
    v16bf a = load_frag_a(A, CN, k0, lane);
#pragma unroll
    for (int t = 0; t < 2; ++t) {
      const int lt = ng * 2 + t;
      v16bf b = load_frag_b(vT + ((long)bidx * CD + head * CDH + lt * 16) * CN,
                            CN, k0, lane);
      acc[t] = wmma_bf16(a, b, acc[t]);
    }
  }
  const int hi = lane >> 4, nn = lane & 15;
#pragma unroll
  for (int t = 0; t < 2; ++t) {
    const int lt = ng * 2 + t;
#pragma unroll
    for (int i = 0; i < 8; ++i) {
      int dd = mt * 16 + i + 8 * hi;
      int ll = lt * 16 + nn;
      attT[((long)bh * CDH + ll) * CDH + dd] = f2bf(acc[t][i]);
    }
  }
}

// ---------------------------------------------------------------------------
// q = xn@Wq + bq, softmax over head dim (64), -> bf16 (BT,512).
// Block: 64 rows x 512 cols, 16 waves (512 thr); wave = 32 rows x one head.
// 2 A-frags reused across 4 B-frags -> 8 WMMA per 6 fragment loads.
// ---------------------------------------------------------------------------
__global__ __launch_bounds__(512) void q_gemm_softmax(
    const u16* __restrict__ xn, const u16* __restrict__ WqT,
    const float* __restrict__ bq, u16* __restrict__ q) {
  const int lane = threadIdx.x & 31, wave = threadIdx.x >> 5;
  const int rhalf = wave & 1, head = wave >> 1;      // head 0..7
  const long row0 = (long)blockIdx.x * 64 + rhalf * 32;
  const int colbase = head * CDH;
  const u16* A0 = xn + row0 * CD;
  const u16* A1 = xn + (row0 + 16) * CD;
  v8f acc[2][4];
#pragma unroll
  for (int rt = 0; rt < 2; ++rt)
#pragma unroll
    for (int t = 0; t < 4; ++t)
#pragma unroll
      for (int i = 0; i < 8; ++i) acc[rt][t][i] = 0.f;

  for (int ks = 0; ks < CD / 32; ++ks) {
    const int k0 = ks * 32;
    v16bf a0 = load_frag_a(A0, CD, k0, lane);
    v16bf a1 = load_frag_a(A1, CD, k0, lane);
    if (k0 + 32 < CD) {   // stream next K-slice of A (global_prefetch_b8)
      __builtin_prefetch(A0 + (lane & 15) * CD + k0 + 32, 0, 0);
      __builtin_prefetch(A1 + (lane & 15) * CD + k0 + 32, 0, 0);
    }
#pragma unroll
    for (int t = 0; t < 4; ++t) {
      v16bf b = load_frag_b(WqT + (long)(colbase + t * 16) * CD, CD, k0, lane);
      acc[0][t] = wmma_bf16(a0, b, acc[0][t]);
      acc[1][t] = wmma_bf16(a1, b, acc[1][t]);
    }
  }
  const int hi = lane >> 4, nn = lane & 15;
#pragma unroll
  for (int t = 0; t < 4; ++t) {
    const float bb = bq[colbase + t * 16 + nn];
#pragma unroll
    for (int rt = 0; rt < 2; ++rt)
#pragma unroll
      for (int i = 0; i < 8; ++i) acc[rt][t][i] += bb;
  }
  // softmax over the 64 head columns of each row (half-wave reductions:
  // C-layout row lives in one VGPR index across the 16 lanes of a half)
#pragma unroll
  for (int rt = 0; rt < 2; ++rt) {
#pragma unroll
    for (int i = 0; i < 8; ++i) {
      float m = fmaxf(fmaxf(acc[rt][0][i], acc[rt][1][i]),
                      fmaxf(acc[rt][2][i], acc[rt][3][i]));
      for (int d = 1; d < 16; d <<= 1) m = fmaxf(m, __shfl_xor(m, d, 32));
      float e[4], s = 0.f;
#pragma unroll
      for (int t = 0; t < 4; ++t) { e[t] = __expf(acc[rt][t][i] - m); s += e[t]; }
      for (int d = 1; d < 16; d <<= 1) s += __shfl_xor(s, d, 32);
      const float inv = 1.f / s;
#pragma unroll
      for (int t = 0; t < 4; ++t) acc[rt][t][i] = e[t] * inv;
    }
  }
#pragma unroll
  for (int rt = 0; rt < 2; ++rt)
#pragma unroll
    for (int t = 0; t < 4; ++t)
#pragma unroll
      for (int i = 0; i < 8; ++i)
        q[(row0 + rt * 16 + i + 8 * hi) * CD + colbase + t * 16 + nn] =
            f2bf(acc[rt][t][i]);
}

// ---------------------------------------------------------------------------
// y = q @ att (per head, K=64), fused: LN over 512 + stylization + SiLU -> h.
// Block: 64 rows x 512 cols, 32 waves; wave = 16 rows x one head.
// ---------------------------------------------------------------------------
__global__ __launch_bounds__(1024) void y_fused(
    const u16* __restrict__ q, const u16* __restrict__ attT,
    const float* __restrict__ pg, const float* __restrict__ pb,
    const float* __restrict__ eo, u16* __restrict__ h) {
  __shared__ float psum[64][CH];
  __shared__ float psqr[64][CH];
  const int lane = threadIdx.x & 31, wave = threadIdx.x >> 5;
  const int rowg = wave & 3, head = wave >> 2;
  const long rowbase = (long)blockIdx.x * 64 + rowg * 16;
  const int b = (int)(blockIdx.x / (CT / 64));   // batch of this row block
  const int colbase = head * CDH;
  v8f acc[4];
#pragma unroll
  for (int t = 0; t < 4; ++t)
#pragma unroll
    for (int i = 0; i < 8; ++i) acc[t][i] = 0.f;

#pragma unroll
  for (int ks = 0; ks < 2; ++ks) {               // K = DH = 64
    const int k0 = ks * 32;
    v16bf a = load_frag_a(q + rowbase * CD + colbase, CD, k0, lane);
#pragma unroll
    for (int t = 0; t < 4; ++t) {
      v16bf bf = load_frag_b(
          attT + ((long)(b * CH + head) * CDH + t * 16) * CDH, CDH, k0, lane);
      acc[t] = wmma_bf16(a, bf, acc[t]);
    }
  }
  const int hi = lane >> 4, nn = lane & 15;
  // per-row partial sums over this wave's 64 columns
#pragma unroll
  for (int i = 0; i < 8; ++i) {
    float s = 0.f, ss = 0.f;
#pragma unroll
    for (int t = 0; t < 4; ++t) { float v = acc[t][i]; s += v; ss += v * v; }
    for (int d = 1; d < 16; d <<= 1) {
      s += __shfl_xor(s, d, 32);
      ss += __shfl_xor(ss, d, 32);
    }
    if (nn == 0) {
      int r = rowg * 16 + i + 8 * hi;
      psum[r][head] = s;
      psqr[r][head] = ss;
    }
  }
  __syncthreads();
#pragma unroll
  for (int i = 0; i < 8; ++i) {
    const int r = rowg * 16 + i + 8 * hi;
    float s = 0.f, ss = 0.f;
#pragma unroll
    for (int hh = 0; hh < CH; ++hh) { s += psum[r][hh]; ss += psqr[r][hh]; }
    const float mean = s * (1.f / (float)CD);
    const float var = ss * (1.f / (float)CD) - mean * mean;
    const float rstd = rsqrtf(var + 1e-5f);
#pragma unroll
    for (int t = 0; t < 4; ++t) {
      const int col = colbase + t * 16 + nn;
      float v = (acc[t][i] - mean) * rstd * pg[col] + pb[col];
      const float sc = eo[(long)b * (2 * CD) + col];
      const float sh = eo[(long)b * (2 * CD) + CD + col];
      v = v * (1.f + sc) + sh;
      v = v / (1.f + __expf(-v));                // SiLU
      h[(rowbase + i + 8 * hi) * CD + col] = f2bf(v);
    }
  }
}

// ---------------------------------------------------------------------------
// out = h @ out_W + out_b + x   (f32 output).
// Block: 64 rows x 512 cols, 16 waves (512 thr); wave = 32 rows x 64 cols.
// ---------------------------------------------------------------------------
__global__ __launch_bounds__(512) void out_gemm(const u16* __restrict__ h,
                                                const u16* __restrict__ WoT,
                                                const float* __restrict__ ob,
                                                const float* __restrict__ x,
                                                float* __restrict__ out) {
  const int lane = threadIdx.x & 31, wave = threadIdx.x >> 5;
  const int rhalf = wave & 1, colg = wave >> 1;      // colg 0..7
  const long row0 = (long)blockIdx.x * 64 + rhalf * 32;
  const int colbase = colg * 64;
  const u16* A0 = h + row0 * CD;
  const u16* A1 = h + (row0 + 16) * CD;
  v8f acc[2][4];
#pragma unroll
  for (int rt = 0; rt < 2; ++rt)
#pragma unroll
    for (int t = 0; t < 4; ++t)
#pragma unroll
      for (int i = 0; i < 8; ++i) acc[rt][t][i] = 0.f;

  for (int ks = 0; ks < CD / 32; ++ks) {
    const int k0 = ks * 32;
    v16bf a0 = load_frag_a(A0, CD, k0, lane);
    v16bf a1 = load_frag_a(A1, CD, k0, lane);
    if (k0 + 32 < CD) {   // stream next K-slice of A (global_prefetch_b8)
      __builtin_prefetch(A0 + (lane & 15) * CD + k0 + 32, 0, 0);
      __builtin_prefetch(A1 + (lane & 15) * CD + k0 + 32, 0, 0);
    }
#pragma unroll
    for (int t = 0; t < 4; ++t) {
      v16bf b = load_frag_b(WoT + (long)(colbase + t * 16) * CD, CD, k0, lane);
      acc[0][t] = wmma_bf16(a0, b, acc[0][t]);
      acc[1][t] = wmma_bf16(a1, b, acc[1][t]);
    }
  }
  const int hi = lane >> 4, nn = lane & 15;
#pragma unroll
  for (int rt = 0; rt < 2; ++rt)
#pragma unroll
    for (int t = 0; t < 4; ++t) {
      const int col = colbase + t * 16 + nn;
      const float bb = ob[col];
#pragma unroll
      for (int i = 0; i < 8; ++i) {
        const long idx = (row0 + rt * 16 + i + 8 * hi) * CD + col;
        out[idx] = acc[rt][t][i] + bb + x[idx];
      }
    }
}

// ---------------------------------------------------------------------------
extern "C" void kernel_launch(void* const* d_in, const int* in_sizes, int n_in,
                              void* d_out, int out_size, void* d_ws,
                              size_t ws_size, hipStream_t stream) {
  const float* x    = (const float*)d_in[0];
  const float* xf   = (const float*)d_in[1];
  const float* emb  = (const float*)d_in[2];
  const int*   cond = (const int*)d_in[3];
  const float* ng   = (const float*)d_in[4];
  const float* nb   = (const float*)d_in[5];
  const float* tg   = (const float*)d_in[6];
  const float* tb   = (const float*)d_in[7];
  const float* Wq   = (const float*)d_in[8];
  const float* bq   = (const float*)d_in[9];
  const float* Wk   = (const float*)d_in[10];
  const float* bk   = (const float*)d_in[11];
  const float* Wv   = (const float*)d_in[12];
  const float* bv   = (const float*)d_in[13];
  const float* eW   = (const float*)d_in[14];
  const float* eb   = (const float*)d_in[15];
  const float* pg   = (const float*)d_in[16];
  const float* pb   = (const float*)d_in[17];
  const float* Wo   = (const float*)d_in[18];
  const float* ob   = (const float*)d_in[19];
  float* out = (float*)d_out;

  char* ws = (char*)d_ws;
  size_t off = 0;
  auto alloc = [&](size_t bytes) -> void* {
    void* p = ws + off;
    off += (bytes + 255) & ~(size_t)255;
    return p;
  };
  u16*   xn   = (u16*)alloc((size_t)CBT * CD * 2);        // 64 MB
  u16*   xfn  = (u16*)alloc((size_t)CB * CN * CL * 2);    // 4 MB
  u16*   qb   = (u16*)alloc((size_t)CBT * CD * 2);        // 64 MB
  u16*   hb   = (u16*)alloc((size_t)CBT * CD * 2);        // 64 MB
  float* kT   = (float*)alloc((size_t)CB * CD * CN * 4);  // 16 MB
  u16*   kTs  = (u16*)alloc((size_t)CB * CD * CN * 2);    // 8 MB
  u16*   vT   = (u16*)alloc((size_t)CB * CD * CN * 2);    // 8 MB
  u16*   attT = (u16*)alloc((size_t)CB * CH * CDH * CDH * 2);
  float* eo   = (float*)alloc((size_t)CB * 2 * CD * 4);
  u16*   WqT  = (u16*)alloc((size_t)CD * CD * 2);
  u16*   WkT  = (u16*)alloc((size_t)CL * CD * 2);
  u16*   WvT  = (u16*)alloc((size_t)CL * CD * 2);
  u16*   WoT  = (u16*)alloc((size_t)CD * CD * 2);

  // 1) weights -> bf16 transposed
  transpose_bf16<<<(CD * CD + 255) / 256, 256, 0, stream>>>(Wq, WqT, CD, CD);
  transpose_bf16<<<(CL * CD + 255) / 256, 256, 0, stream>>>(Wk, WkT, CL, CD);
  transpose_bf16<<<(CL * CD + 255) / 256, 256, 0, stream>>>(Wv, WvT, CL, CD);
  transpose_bf16<<<(CD * CD + 255) / 256, 256, 0, stream>>>(Wo, WoT, CD, CD);

  // 2) LayerNorms -> bf16
  ln_rows<<<CBT / 8, 256, 0, stream>>>(x, ng, nb, xn, CD);
  ln_rows<<<(CB * CN) / 8, 256, 0, stream>>>(xf, tg, tb, xfn, CL);

  // 3) emb MLP (scale/shift)
  emb_gemm<<<CB, 256, 0, stream>>>(emb, eW, eb, eo);

  // 4) k and v projections (WMMA) with mask / tct epilogues
  kv_gemm<<<CB * (CN / 64), 1024, 0, stream>>>(xfn, WkT, bk, cond, kT, vT, 0);
  kv_gemm<<<CB * (CN / 64), 1024, 0, stream>>>(xfn, WvT, bv, cond, kT, vT, 1);

  // 5) softmax over N, then att = k^T v (WMMA)
  softmax_rows<<<(CB * CD) / 8, 256, 0, stream>>>(kT, kTs);
  att_gemm<<<CB * CH, 256, 0, stream>>>(kTs, vT, attT);

  // 6) q projection + head softmax (WMMA)
  q_gemm_softmax<<<CBT / 64, 512, 0, stream>>>(xn, WqT, bq, qb);

  // 7) y = q@att + LN + stylization + SiLU (WMMA)
  y_fused<<<CBT / 64, 1024, 0, stream>>>(qb, attT, pg, pb, eo, hb);

  // 8) output projection + residual (WMMA)
  out_gemm<<<CBT / 64, 512, 0, stream>>>(hb, WoT, ob, x, out);
}
